// DeformConv_90288802496925
// MI455X (gfx1250) — matrix operand order repaired and checked by reading, not compiled, not run on hardware
//
#include <hip/hip_runtime.h>

typedef __attribute__((ext_vector_type(16))) _Float16 v16h;
typedef __attribute__((ext_vector_type(8)))  float    v8f;

#define Bn    4
#define Cin   64
#define Hh    128
#define Ww    128
#define CoutN 64
#define KK    9
#define HW    (Hh * Ww)        // 16384
#define CK    576              // Cin * KK
#define OFFC  18               // 2*K offset channels
#define SROW  592              // 576 + pad, 592*2 = 1184 B = 37*32 -> 32B-aligned rows

__device__ __forceinline__ int iclamp(int v, int lo, int hi) {
    return v < lo ? lo : (v > hi ? hi : v);
}

// ---------------------------------------------------------------------------
// Kernel 0: convert weight (O=64, C=64, 3,3) f32 -> f16 WMMA A-fragments in ws
// Fragment layout: contiguous k = 16*(lane/16) + j, same order as B; k-major contraction index
// ck = kk*64 + c.  Storage: halfs[((t*18 + ks)*32 + lane)*16 + j]
//   lane<16 :  K=ks*32+j
//   lane>=16:  K=ks*32+16+j
//   M (cout within tile) = lane & 15
// ---------------------------------------------------------------------------
__global__ __launch_bounds__(256) void weight_prep_kernel(
    const float* __restrict__ weight, _Float16* __restrict__ wp)
{
    int i = blockIdx.x * 256 + threadIdx.x;      // 0 .. 4*18*32*16-1 = 36863
    if (i >= 4 * 18 * 32 * 16) return;
    int j    = i & 15;
    int lane = (i >> 4) & 31;
    int ks   = (i >> 9) % 18;
    int t    = (i >> 9) / 18;
    int K    = ks * 32 + ((lane < 16) ? j : 16 + j);
    int kk   = K >> 6;            // k-major: ck = kk*64 + c
    int c    = K & 63;
    int cout = t * 16 + (lane & 15);
    wp[i] = (_Float16)weight[cout * CK + c * KK + kk];   // (O,C,3,3) flat
}

// ---------------------------------------------------------------------------
// Kernel 1: offsets = relu(conv3x3_same(x, w_off, b_off)) -> d_out tail
// One thread per pixel, w_off (18*576 f32 = 41.5 KB) staged in LDS.
// ---------------------------------------------------------------------------
__global__ __launch_bounds__(256) void offsets_conv_kernel(
    const float* __restrict__ x, const float* __restrict__ w_off,
    const float* __restrict__ b_off, float* __restrict__ offs)
{
    __shared__ float wlds[OFFC * CK];
    const int tid = threadIdx.x;
    for (int i = tid; i < OFFC * CK; i += 256) wlds[i] = w_off[i];
    __syncthreads();

    int p = blockIdx.x * 256 + tid;
    int b = p >> 14, h = (p >> 7) & 127, w = p & 127;

    float acc[OFFC];
#pragma unroll
    for (int o = 0; o < OFFC; ++o) acc[o] = b_off[o];

    const float* xb = x + b * Cin * HW;
    for (int ky = 0; ky < 3; ++ky) {
        int yy = h - 1 + ky;
        if (yy < 0 || yy >= Hh) continue;
        for (int kx = 0; kx < 3; ++kx) {
            int xx = w - 1 + kx;
            if (xx < 0 || xx >= Ww) continue;
            const float* xp = xb + yy * Ww + xx;
            const float* wq = wlds + ky * 3 + kx;   // broadcast LDS reads
            for (int c = 0; c < Cin; ++c) {
                float xv = xp[c * HW];
#pragma unroll
                for (int o = 0; o < OFFC; ++o)
                    acc[o] = fmaf(xv, wq[o * CK + c * KK], acc[o]);
            }
        }
    }
    float* op = offs + b * OFFC * HW + h * Ww + w;
#pragma unroll
    for (int o = 0; o < OFFC; ++o) op[o * HW] = fmaxf(acc[o], 0.0f);
}

// ---------------------------------------------------------------------------
// Kernel 2: deformable conv via f16 WMMA.
// Block = 32 pixels (one h-row strip) x all 64 couts; 256 threads = 8 waves.
// Stage 1: bilinear-sample 576x32 f16 activation tile into LDS (k-major).
// Stage 2: wave (t, pt) computes 16x16 output tile with 18x
//          v_wmma_f32_16x16x32_f16 over K=576.
// ---------------------------------------------------------------------------
__global__ __launch_bounds__(256) void deform_wmma_kernel(
    const float* __restrict__ x, const float* __restrict__ offs,
    const _Float16* __restrict__ wprep, float* __restrict__ out)
{
    __shared__ _Float16 s[32 * SROW];           // 37,888 B

    const int tid = threadIdx.x;
    const int blk = blockIdx.x;                 // 2048 blocks
    const int b   = blk >> 9;
    const int h   = (blk >> 2) & 127;
    const int w0  = (blk & 3) << 5;

    // ---- Stage 1: gather + bilinear into LDS -----------------------------
    const float* xb = x + b * Cin * HW;
    for (int task = tid; task < 32 * KK; task += 256) {
        int kk  = task >> 5;                    // 0..8 (uniform per wave pass)
        int pix = task & 31;
        int w   = w0 + pix;
        int ky  = kk / 3, kx = kk - ky * 3;

        const float* ob = offs + (b * OFFC + kk * 2) * HW + h * Ww + w;
        float dy = ob[0];
        float dx = ob[HW];
        float py = (float)(h - 1 + ky) + dy;
        float px = (float)(w - 1 + kx) + dx;
        float fy = floorf(py), fx = floorf(px);
        float wy = py - fy,    wx = px - fx;
        int y0 = (int)fy, x0 = (int)fx;
        int y1 = y0 + 1,  x1 = x0 + 1;

        float vy0 = (y0 >= 0 && y0 < Hh) ? 1.0f : 0.0f;
        float vy1 = (y1 >= 0 && y1 < Hh) ? 1.0f : 0.0f;
        float vx0 = (x0 >= 0 && x0 < Ww) ? 1.0f : 0.0f;
        float vx1 = (x1 >= 0 && x1 < Ww) ? 1.0f : 0.0f;
        float w00 = (1.0f - wy) * (1.0f - wx) * vy0 * vx0;
        float w01 = (1.0f - wy) * wx          * vy0 * vx1;
        float w10 = wy * (1.0f - wx)          * vy1 * vx0;
        float w11 = wy * wx                   * vy1 * vx1;

        int y0c = iclamp(y0, 0, Hh - 1), y1c = iclamp(y1, 0, Hh - 1);
        int x0c = iclamp(x0, 0, Ww - 1), x1c = iclamp(x1, 0, Ww - 1);
        int a00 = y0c * Ww + x0c, a01 = y0c * Ww + x1c;
        int a10 = y1c * Ww + x0c, a11 = y1c * Ww + x1c;

        _Float16* sp = s + pix * SROW + kk * 64;   // k-major: ck = kk*64 + c
        const float* xc = xb;
        for (int c = 0; c < Cin; ++c, xc += HW) {
            float v = w00 * xc[a00] + w01 * xc[a01]
                    + w10 * xc[a10] + w11 * xc[a11];
            sp[c] = (_Float16)v;
        }
    }
    __syncthreads();

    // ---- Stage 2: WMMA over K=576 (18 steps of 32) -----------------------
    const int wave = tid >> 5, lane = tid & 31;
    const int t    = wave & 3;         // cout tile: couts [t*16, t*16+16)
    const int pt   = wave >> 2;        // pixel tile: pixels [pt*16, pt*16+16)

    v8f acc = {};
    // B fragment: lane<16 -> K 0..15 of column N=lane; lane>=16 -> K 16..31
    const _Float16* srow  = s + (pt * 16 + (lane & 15)) * SROW + ((lane >> 4) << 4);
    // A fragment: prepped contiguous 16 halfs per lane per k-step
    const _Float16* wlane = wprep + (t * 18 * 32 + lane) * 16;

#pragma unroll
    for (int ks = 0; ks < 18; ++ks) {
        v16h a  = *(const v16h*)(wlane + ks * 512);   // 32 B global load
        v16h bf = *(const v16h*)(srow  + ks * 32);    // 32 B LDS load
        acc = __builtin_amdgcn_wmma_f32_16x16x32_f16(
                  false, a, false, bf, (short)0, acc, false, false);
    }

    // D layout: VGPR r, lane l -> M = r + 8*(l>=16), N = l&15
    const int coutBase = t * 16 + ((lane >> 4) << 3);
    const int pix      = pt * 16 + (lane & 15);
    float* op = out + (b * CoutN) * HW + h * Ww + (w0 + pix);
#pragma unroll
    for (int r = 0; r < 8; ++r)
        op[(coutBase + r) * HW] = acc[r];
}

// ---------------------------------------------------------------------------
extern "C" void kernel_launch(void* const* d_in, const int* in_sizes, int n_in,
                              void* d_out, int out_size, void* d_ws, size_t ws_size,
                              hipStream_t stream)
{
    const float* x      = (const float*)d_in[0];   // (4,64,128,128)
    const float* w_off  = (const float*)d_in[1];   // (18,64,3,3)
    const float* b_off  = (const float*)d_in[2];   // (18,)
    const float* weight = (const float*)d_in[3];   // (64,64,3,3)

    float* out  = (float*)d_out;                   // (4,64,128,128)
    float* offs = out + Bn * CoutN * HW;           // tuple tail: (4,18,128,128)
    _Float16* wprep = (_Float16*)d_ws;             // 73,728 B of fragments

    weight_prep_kernel <<<144,  256, 0, stream>>>(weight, wprep);
    offsets_conv_kernel<<<256,  256, 0, stream>>>(x, w_off, b_off, offs);
    deform_wmma_kernel <<<2048, 256, 0, stream>>>(x, offs, wprep, out);
}
